// ConditionalSelection_58514634440931
// MI455X (gfx1250) — matrix operand order repaired and checked by reading, not compiled, not run on hardware
//
#include <hip/hip_runtime.h>
#include <hip/hip_bf16.h>

typedef _Float16 h16;
typedef __attribute__((ext_vector_type(8)))  _Float16 v8h;
typedef __attribute__((ext_vector_type(16))) _Float16 v16h;
typedef __attribute__((ext_vector_type(8)))  float    v8f;

static __device__ __forceinline__ v16h cat8(v8h lo, v8h hi) {
    return __builtin_shufflevector(lo, hi, 0,1,2,3,4,5,6,7,8,9,10,11,12,13,14,15);
}

// ---------------------------------------------------------------------------
// f32 -> f16 convert (grid-stride)
// ---------------------------------------------------------------------------
__global__ void cvt_f32_f16(const float* __restrict__ in, h16* __restrict__ out, long n) {
    long i = (long)blockIdx.x * blockDim.x + threadIdx.x;
    long stride = (long)gridDim.x * blockDim.x;
    for (; i < n; i += stride) out[i] = (h16)in[i];
}

// ---------------------------------------------------------------------------
// WMMA GEMM: C[M,N] = A[M,K] @ W[N,K]^T + bias
//   A: f16 row-major with row stride lda (halves), KC multiple of 32
//   W: f16 row-major [N,K]
// One wave32 computes a 32(M) x 64(N) block = 8 accumulator tiles (2 Mx4 N).
// KC is compile-time; full unroll lets the scheduler overlap loads and WMMAs.
// MODE 0: store f16 row-major, ld = N  (used for h and qkv)
// MODE 1: store f32 permuted into d_out: row=3b+t -> out[t*Btot*512 + b*512 + col]
// ---------------------------------------------------------------------------
template<int MODE, int KC>
__global__ __launch_bounds__(256)
void gemm_wmma_f16(const h16* __restrict__ A, int lda,
                   const h16* __restrict__ W, const float* __restrict__ bias,
                   h16* __restrict__ outH, float* __restrict__ outF,
                   int M, int N, int Btot)
{
    const int lane = threadIdx.x & 31;
    const int wave = threadIdx.x >> 5;
    const int nTiles = N >> 6;                      // 64-wide N blocks
    const long gw = (long)blockIdx.x * 8 + wave;    // global wave index
    const int m32  = (int)(gw / nTiles);
    const int nblk = (int)(gw % nTiles);
    const int mbase = m32 << 5;
    const int n0    = nblk << 6;
    const int l = lane & 15;
    const int g = lane >> 4;

    // A fragments: lane holds row (mbase + 16*mt + l); K chunks at k0+8g, k0+16+8g
    const h16* Ab[2];
#pragma unroll
    for (int mt = 0; mt < 2; ++mt)
        Ab[mt] = A + (size_t)(mbase + (mt << 4) + l) * (size_t)lda + (g << 3);
    // B fragments (tile t): lane holds col n0+16t+l; K chunks at k0+8g, k0+16+8g
    const h16* Wb[4];
#pragma unroll
    for (int t = 0; t < 4; ++t)
        Wb[t] = W + (size_t)(n0 + (t << 4) + l) * (size_t)KC + (g << 3);

    v8f acc[8];                                      // acc[mt*4 + t]
#pragma unroll
    for (int q = 0; q < 8; ++q)
#pragma unroll
        for (int i = 0; i < 8; ++i) acc[q][i] = 0.0f;

#pragma unroll
    for (int k0 = 0; k0 < KC; k0 += 32) {
        v16h a[2];
#pragma unroll
        for (int mt = 0; mt < 2; ++mt) {
            v8h a0 = *(const v8h*)(Ab[mt] + k0);
            v8h a1 = *(const v8h*)(Ab[mt] + k0 + 16);
            a[mt] = cat8(a0, a1);
        }
#pragma unroll
        for (int t = 0; t < 4; ++t) {
            v8h b0 = *(const v8h*)(Wb[t] + k0);
            v8h b1 = *(const v8h*)(Wb[t] + k0 + 16);
            v16h b = cat8(b0, b1);
#pragma unroll
            for (int mt = 0; mt < 2; ++mt)
                acc[mt * 4 + t] = __builtin_amdgcn_wmma_f32_16x16x32_f16(
                    false, a[mt], false, b, (short)0, acc[mt * 4 + t], false, false);
        }
    }

    // Epilogue. C/D layout: VGPR r -> M = mbase + 16*mt + 8*g + r, N = n0 + 16*t + l
#pragma unroll
    for (int t = 0; t < 4; ++t) {
        const int col = n0 + (t << 4) + l;
        const float bv = bias[col];
#pragma unroll
        for (int mt = 0; mt < 2; ++mt) {
#pragma unroll
            for (int r = 0; r < 8; ++r) {
                const int row = mbase + (mt << 4) + (g << 3) + r;
                const float val = acc[mt * 4 + t][r] + bv;
                if (MODE == 0) {
                    outH[(size_t)row * (size_t)N + col] = (h16)val;
                } else {
                    const int bb = row / 3;
                    const int tt = row - 3 * bb;
                    outF[(size_t)tt * ((size_t)Btot * 512) + (size_t)bb * 512 + col] = val;
                }
            }
        }
    }
}

// ---------------------------------------------------------------------------
// In-place LayerNorm(1536) + ReLU over f16 rows. One 256-thread block per row.
// ---------------------------------------------------------------------------
__global__ __launch_bounds__(256)
void ln_relu_kernel(h16* __restrict__ h, const float* __restrict__ gam,
                    const float* __restrict__ bet)
{
    const int tid = threadIdx.x;
    h16* rp = h + (size_t)blockIdx.x * 1536;

    float v[6], s = 0.f, s2 = 0.f;
#pragma unroll
    for (int i = 0; i < 6; ++i) {
        const int c = tid + (i << 8);
        v[i] = (float)rp[c];
        s += v[i];
        s2 += v[i] * v[i];
    }
    __shared__ float r1[256], r2[256];
    r1[tid] = s; r2[tid] = s2;
    __syncthreads();
    for (int st = 128; st > 0; st >>= 1) {
        if (tid < st) { r1[tid] += r1[tid + st]; r2[tid] += r2[tid + st]; }
        __syncthreads();
    }
    const float mean = r1[0] * (1.0f / 1536.0f);
    const float var  = r2[0] * (1.0f / 1536.0f) - mean * mean;
    const float rstd = rsqrtf(var + 1e-5f);
#pragma unroll
    for (int i = 0; i < 6; ++i) {
        const int c = tid + (i << 8);
        float y = (v[i] - mean) * rstd * gam[c] + bet[c];
        y = fmaxf(y, 0.0f);
        rp[c] = (h16)y;
    }
}

// ---------------------------------------------------------------------------
// 3x3 attention per (batch, head). qkv row layout [3B,1536] = [q|k|v].
// ctx is written in place over the q slice.
// ---------------------------------------------------------------------------
__global__ __launch_bounds__(256)
void attn_kernel(h16* __restrict__ qkv, int Btot)
{
    const long idx = (long)blockIdx.x * blockDim.x + threadIdx.x;
    if (idx >= (long)Btot * 8) return;
    const int b  = (int)(idx >> 3);
    const int hd = (int)(idx & 7);
    h16* base = qkv + (size_t)b * 3 * 1536 + hd * 64;

    float sc[3][3];
#pragma unroll
    for (int i = 0; i < 3; ++i)
#pragma unroll
        for (int j = 0; j < 3; ++j) sc[i][j] = 0.0f;

    for (int d8 = 0; d8 < 8; ++d8) {
        v8h q[3], k[3];
#pragma unroll
        for (int t = 0; t < 3; ++t) {
            q[t] = *(const v8h*)(base + t * 1536 + d8 * 8);
            k[t] = *(const v8h*)(base + t * 1536 + 512 + d8 * 8);
        }
#pragma unroll
        for (int ti = 0; ti < 3; ++ti)
#pragma unroll
            for (int si = 0; si < 3; ++si)
#pragma unroll
                for (int i = 0; i < 8; ++i)
                    sc[ti][si] += (float)q[ti][i] * (float)k[si][i];
    }

    float p[3][3];
#pragma unroll
    for (int t = 0; t < 3; ++t) {
        const float a0 = sc[t][0] * 0.125f, a1 = sc[t][1] * 0.125f, a2 = sc[t][2] * 0.125f;
        const float m = fmaxf(a0, fmaxf(a1, a2));
        const float e0 = __expf(a0 - m), e1 = __expf(a1 - m), e2 = __expf(a2 - m);
        const float inv = 1.0f / (e0 + e1 + e2);
        p[t][0] = e0 * inv; p[t][1] = e1 * inv; p[t][2] = e2 * inv;
    }

    for (int d8 = 0; d8 < 8; ++d8) {
        v8h vv[3];
#pragma unroll
        for (int t = 0; t < 3; ++t)
            vv[t] = *(const v8h*)(base + t * 1536 + 1024 + d8 * 8);
#pragma unroll
        for (int t = 0; t < 3; ++t) {
            v8h o;
#pragma unroll
            for (int i = 0; i < 8; ++i) {
                const float a = p[t][0] * (float)vv[0][i]
                              + p[t][1] * (float)vv[1][i]
                              + p[t][2] * (float)vv[2][i];
                o[i] = (h16)a;
            }
            *(v8h*)(base + t * 1536 + d8 * 8) = o;   // overwrite q slice
        }
    }
}

// ---------------------------------------------------------------------------
// Gumbel softmax over the 3 token slices, in place in d_out.
// ---------------------------------------------------------------------------
__global__ __launch_bounds__(256)
void gumbel_softmax_kernel(float* __restrict__ out, const float* __restrict__ gum, int Btot)
{
    const long idx = (long)blockIdx.x * blockDim.x + threadIdx.x;
    const long total = (long)Btot * 512;
    if (idx >= total) return;
    const int b = (int)(idx >> 9);
    const int c = (int)(idx & 511);
    const size_t S = (size_t)Btot * 512;
    const size_t gb = (size_t)b * 1536 + c;

    const float a0 = out[idx]         + gum[gb];
    const float a1 = out[S + idx]     + gum[gb + 512];
    const float a2 = out[2 * S + idx] + gum[gb + 1024];
    const float m = fmaxf(a0, fmaxf(a1, a2));
    const float e0 = __expf(a0 - m), e1 = __expf(a1 - m), e2 = __expf(a2 - m);
    const float inv = 1.0f / (e0 + e1 + e2);
    out[idx]         = e0 * inv;
    out[S + idx]     = e1 * inv;
    out[2 * S + idx] = e2 * inv;
}

// ---------------------------------------------------------------------------
extern "C" void kernel_launch(void* const* d_in, const int* in_sizes, int n_in,
                              void* d_out, int out_size, void* d_ws, size_t ws_size,
                              hipStream_t stream)
{
    const float* x    = (const float*)d_in[0];
    const float* W_fc = (const float*)d_in[1];
    const float* b_fc = (const float*)d_in[2];
    const float* ln_g = (const float*)d_in[3];
    const float* ln_b = (const float*)d_in[4];
    const float* Wqkv = (const float*)d_in[5];
    const float* bqkv = (const float*)d_in[6];
    const float* Wo   = (const float*)d_in[7];
    const float* bo   = (const float*)d_in[8];
    const float* gum  = (const float*)d_in[9];
    float* out = (float*)d_out;

    const int B = in_sizes[0] / 512;   // 65536
    const int H3 = 1536;
    constexpr int K = 512;

    // workspace layout (256B aligned chunks)
    char* ws = (char*)d_ws;
    size_t off = 0;
    auto take = [&](size_t bytes) { void* p = ws + off; off += (bytes + 255) & ~(size_t)255; return p; };
    h16* xf16   = (h16*)take((size_t)B * 512 * 2);
    h16* wfc16  = (h16*)take((size_t)H3 * K * 2);
    h16* wqkv16 = (h16*)take((size_t)H3 * K * 2);
    h16* wo16   = (h16*)take((size_t)512 * K * 2);
    h16* hact   = (h16*)take((size_t)B * 1536 * 2);          // h, then LN+ReLU in place
    h16* qkv    = (h16*)take((size_t)B * 3 * 1536 * 2);      // qkv, ctx in place over q
    (void)ws_size; (void)n_in; (void)out_size;

    // 1) converts
    cvt_f32_f16<<<4096, 256, 0, stream>>>(x,    xf16,   (long)B * 512);
    cvt_f32_f16<<<1024, 256, 0, stream>>>(W_fc, wfc16,  (long)H3 * K);
    cvt_f32_f16<<<1024, 256, 0, stream>>>(Wqkv, wqkv16, (long)H3 * K);
    cvt_f32_f16<<<512,  256, 0, stream>>>(Wo,   wo16,   (long)512 * K);

    // 2) GEMM1: h = x @ W_fc^T + b_fc  -> hact f16 [B,1536]
    {
        const int waves = (B / 32) * (H3 / 64);
        gemm_wmma_f16<0, K><<<waves / 8, 256, 0, stream>>>(
            xf16, 512, wfc16, b_fc, hact, nullptr, B, H3, B);
    }

    // 3) LayerNorm + ReLU in place
    ln_relu_kernel<<<B, 256, 0, stream>>>(hact, ln_g, ln_b);

    // 4) GEMM2: qkv = x3 @ Wqkv^T + bqkv ; x3 = hact viewed [3B,512]
    {
        const int M = 3 * B;
        const int waves = (M / 32) * (H3 / 64);
        gemm_wmma_f16<0, K><<<waves / 8, 256, 0, stream>>>(
            hact, 512, wqkv16, bqkv, qkv, nullptr, M, H3, B);
    }

    // 5) attention (ctx written over the q slice of qkv)
    attn_kernel<<<(B * 8) / 256, 256, 0, stream>>>(qkv, B);

    // 6) GEMM3: attn_out = ctx @ Wo^T + bo -> f32 into d_out as [t][b][c]
    {
        const int M = 3 * B;
        const int waves = (M / 32) * (512 / 64);
        gemm_wmma_f16<1, K><<<waves / 8, 256, 0, stream>>>(
            qkv /* q slice = ctx */, 1536, wo16, bo, nullptr, out, M, 512, B);
    }

    // 7) gumbel softmax over tokens, in place in d_out
    gumbel_softmax_kernel<<<(B * 512) / 256, 256, 0, stream>>>(out, gum, B);
}
